// mixWeightNN_10522669875817
// MI455X (gfx1250) — matrix operand order repaired and checked by reading, not compiled, not run on hardware
//
#include <hip/hip_runtime.h>

typedef __attribute__((ext_vector_type(16))) __bf16 v16bf;
typedef __attribute__((ext_vector_type(8)))  float  v8f;

// Problem dims (from reference setup_inputs)
constexpr int Bdim = 4096;
constexpr int Edim = 8;
constexpr int Mdim = 1024;
constexpr int Ndim = 1024;
constexpr int Kdim = Edim * Mdim; // 8192 fused contraction dim

// Tiling
constexpr int BM = 128;
constexpr int BN = 128;
constexpr int BK = 32;
constexpr int TILE_A = BM * BK;   // 4096 halfs per (hi|lo) buffer
constexpr int TILE_B = BK * BN;   // 4096 halfs per (hi|lo) buffer
constexpr int KT = Kdim / BK;     // 256 K-steps
constexpr int NBLK = Ndim / 16;   // 64 column blocks in packed-W layout

// Pack the high 16 bits (truncated bf16) of two floats into one dword.
__device__ __forceinline__ unsigned pack_hi16(float f0, float f1) {
    return __builtin_amdgcn_perm(__float_as_uint(f1), __float_as_uint(f0), 0x07060302u);
}

// Exact remainder after removing the truncated-bf16 high part.
__device__ __forceinline__ float bf16_rem(float v) {
    return v - __uint_as_float(__float_as_uint(v) & 0xFFFF0000u);
}

__device__ __forceinline__ void wait_asynccnt0() {
#if __has_builtin(__builtin_amdgcn_s_wait_asynccnt)
    __builtin_amdgcn_s_wait_asynccnt(0);
#else
    asm volatile("s_wait_asynccnt 0x0" ::: "memory");
#endif
}

// ============================================================================
// Pack kernel: W [K,N] fp32 -> Whi/Wlo bf16 in WMMA-B fragment order:
//   frag(kblk, nblk): lane L holds col n = nblk*16+(L&15), K(h) = (L>>4)*16+h
//   half index = ((kblk*NBLK + nblk)*32 + lane)*16 + h
// ============================================================================
__global__ __launch_bounds__(256) void pack_w(const float* __restrict__ W,
                                              unsigned short* __restrict__ Whi,
                                              unsigned short* __restrict__ Wlo)
{
    const int idx = blockIdx.x * 256 + threadIdx.x;  // k-pair id: 4M total
    const int n   = idx & (Ndim - 1);                // column
    const int k   = (idx >> 10) << 1;                // even k
    const float v0 = W[k * Ndim + n];
    const float v1 = W[(k + 1) * Ndim + n];
    const int kblk = k >> 5;
    const int kk   = k & 31;
    const int nblk = n >> 4;
    const int lane = (n & 15) | ((kk >> 4) << 4);
    const int off  = ((kblk * NBLK + nblk) * 32 + lane) * 16 + (kk & 15); // even
    *(unsigned int*)&Whi[off] = pack_hi16(v0, v1);
    *(unsigned int*)&Wlo[off] = pack_hi16(bf16_rem(v0), bf16_rem(v1));
}

// ============================================================================
// Fast path GEMM: A (ew-scaled x) via LDS pipeline; B fragments DMA'd
// global->LDS with GLOBAL_LOAD_ASYNC_TO_LDS_B128 (ASYNCcnt), double-buffered.
// ============================================================================

struct StageA {
    float4 a[4];    // 16 x-elements (4 quads of 4 consecutive k)
    float  sw[4];   // ew scale per quad's row
};

struct OffsetsA {
    int aoff[4];    // r*Mdim + kq  (A global, floats)
    int eoff[4];    // r*Edim       (ew global, floats)
    int offA[4];    // LDS half-index for A quad (8B aligned)
};

__device__ __forceinline__ void init_offsetsA(int tid, OffsetsA& o) {
    #pragma unroll
    for (int j = 0; j < 4; ++j) {
        const int q  = j * 256 + tid;     // quad index 0..1023
        const int r  = q >> 3;            // tile row 0..127
        const int kq = (q & 7) << 2;      // kk0: 0,4,...,28
        o.aoff[j] = r * Mdim + kq;
        o.eoff[j] = r * Edim;
        const int lnA = (r & 15) | ((kq >> 4) << 4);
        const int hhA = kq & 15;
        o.offA[j] = ((r >> 4) * 512) + lnA * 16 + hhA;
    }
}

__device__ __forceinline__ void load_stageA(const float* __restrict__ x,
                                            const float* __restrict__ ew,
                                            int bm0, int kt,
                                            const OffsetsA& o, StageA& s)
{
    const int k0 = kt * BK;
    const int e  = k0 >> 10;               // expert index (uniform)
    const int m0 = k0 & (Mdim - 1);        // m within expert (uniform)
    const float* __restrict__ xb  = x  + bm0 * Mdim + m0;   // uniform base
    const float* __restrict__ ewb = ew + bm0 * Edim + e;    // uniform base
    #pragma unroll
    for (int j = 0; j < 4; ++j) {
        s.a[j]  = *(const float4*)(xb + o.aoff[j]);
        s.sw[j] = ewb[o.eoff[j]];
    }
}

__device__ __forceinline__ void stageA_to_lds(const StageA& s, const OffsetsA& o,
                                              unsigned short* __restrict__ As_hi,
                                              unsigned short* __restrict__ As_lo)
{
    #pragma unroll
    for (int j = 0; j < 4; ++j) {
        const float sw = s.sw[j];
        const float v0 = s.a[j].x * sw, v1 = s.a[j].y * sw;
        const float v2 = s.a[j].z * sw, v3 = s.a[j].w * sw;
        uint2 ph, pl;
        ph.x = pack_hi16(v0, v1);
        ph.y = pack_hi16(v2, v3);
        pl.x = pack_hi16(bf16_rem(v0), bf16_rem(v1));
        pl.y = pack_hi16(bf16_rem(v2), bf16_rem(v3));
        *(uint2*)&As_hi[o.offA[j]] = ph;   // 8B-aligned b64 store
        *(uint2*)&As_lo[o.offA[j]] = pl;
    }
}

// Async DMA one 8KB hi (or lo) B slice (contiguous, fragment-ordered) into LDS.
// Each lane moves 16B per instruction; 2 instructions cover 256 threads * 16B * 2
// = 8KB. INST_OFFSET is added to BOTH the LDS and memory addresses (ISA 10.x).
__device__ __forceinline__ void async_copy_B(const unsigned short* __restrict__ src,
                                             unsigned lds_base_bytes, int tid)
{
    const unsigned voff = (unsigned)(tid << 4);      // 16B per lane
    const unsigned dst  = lds_base_bytes + voff;     // LDS byte address
    asm volatile("global_load_async_to_lds_b128 %0, %1, %2"
                 :: "v"(dst), "v"(voff), "s"(src) : "memory");
    asm volatile("global_load_async_to_lds_b128 %0, %1, %2 offset:4096"
                 :: "v"(dst), "v"(voff), "s"(src) : "memory");
}

// 24 WMMAs: 2x4 16x16 tiles per wave; B fragments read from LDS.
__device__ __forceinline__ void compute_tile_ws(const unsigned short* __restrict__ As_hi,
                                                const unsigned short* __restrict__ As_lo,
                                                const unsigned short* __restrict__ Bs_hi,
                                                const unsigned short* __restrict__ Bs_lo,
                                                int lane, int wave_m, int wave_n,
                                                v8f acc[2][4])
{
    v16bf ah[2], al[2];
    #pragma unroll
    for (int i = 0; i < 2; ++i) {
        const int rb = wave_m * 2 + i;
        ah[i] = *(const v16bf*)&As_hi[rb * 512 + lane * 16];
        al[i] = *(const v16bf*)&As_lo[rb * 512 + lane * 16];
    }
    #pragma unroll
    for (int jn = 0; jn < 4; ++jn) {
        const int fo = (wave_n * 4 + jn) * 512 + lane * 16;
        const v16bf bhi = *(const v16bf*)&Bs_hi[fo];
        const v16bf blo = *(const v16bf*)&Bs_lo[fo];
        #pragma unroll
        for (int i = 0; i < 2; ++i) {
            acc[i][jn] = __builtin_amdgcn_wmma_f32_16x16x32_bf16(
                false, ah[i], false, bhi, (short)0, acc[i][jn], false, false);
            acc[i][jn] = __builtin_amdgcn_wmma_f32_16x16x32_bf16(
                false, ah[i], false, blo, (short)0, acc[i][jn], false, false);
            acc[i][jn] = __builtin_amdgcn_wmma_f32_16x16x32_bf16(
                false, al[i], false, bhi, (short)0, acc[i][jn], false, false);
        }
    }
}

__global__ __launch_bounds__(256) void moe_mix_gemm_ws(
    const float* __restrict__ x,              // [B, M]
    const float* __restrict__ ew,             // [B, E]
    const unsigned short* __restrict__ Whi,   // packed B fragments (hi)
    const unsigned short* __restrict__ Wlo,   // packed B fragments (lo)
    const float* __restrict__ bias,           // [N]
    float* __restrict__ out)                  // [B, N]
{
    __shared__ alignas(32) unsigned short As_hi[2][TILE_A];   // 16 KB
    __shared__ alignas(32) unsigned short As_lo[2][TILE_A];   // 16 KB
    __shared__ alignas(32) unsigned short Bs_hi[2][TILE_B];   // 16 KB (async dest)
    __shared__ alignas(32) unsigned short Bs_lo[2][TILE_B];   // 16 KB (async dest)

    const int tid    = threadIdx.x;
    const int lane   = tid & 31;
    const int wid    = tid >> 5;      // 0..7
    const int wave_m = wid & 3;       // 4 row groups of 32
    const int wave_n = wid >> 2;      // 2 col groups of 64

    const int bm0    = blockIdx.x * BM;
    const int bn0    = blockIdx.y * BN;
    const int bn_blk = bn0 >> 4;      // first of this WG's 8 column blocks

    // LDS byte addresses for async destinations (low 32 bits of generic addr).
    const unsigned bhi_lds[2] = { (unsigned)(uintptr_t)&Bs_hi[0][0],
                                  (unsigned)(uintptr_t)&Bs_hi[1][0] };
    const unsigned blo_lds[2] = { (unsigned)(uintptr_t)&Bs_lo[0][0],
                                  (unsigned)(uintptr_t)&Bs_lo[1][0] };

    OffsetsA o;
    init_offsetsA(tid, o);

    v8f acc[2][4] = {};
    StageA s;

    // ---- Pipeline prologue ----
    async_copy_B(Whi + ((size_t)0 * NBLK + bn_blk) * 512, bhi_lds[0], tid);
    async_copy_B(Wlo + ((size_t)0 * NBLK + bn_blk) * 512, blo_lds[0], tid);
    load_stageA(x, ew, bm0, 0, o, s);
    stageA_to_lds(s, o, As_hi[0], As_lo[0]);
    load_stageA(x, ew, bm0, 1, o, s);
    wait_asynccnt0();
    __syncthreads();

    // ---- Main loop: 1 barrier per K-step ----
    for (int t = 0; t < KT - 2; ++t) {
        const int nxt = (t + 1) & 1;
        const int cur = t & 1;
        // A tile t+1: regs -> alternate LDS buffer
        stageA_to_lds(s, o, As_hi[nxt], As_lo[nxt]);
        // B tile t+1: async DMA into alternate LDS buffer (overlaps compute)
        async_copy_B(Whi + ((size_t)(t + 1) * NBLK + bn_blk) * 512, bhi_lds[nxt], tid);
        async_copy_B(Wlo + ((size_t)(t + 1) * NBLK + bn_blk) * 512, blo_lds[nxt], tid);
        // A tile t+2 global loads (latency hidden across compute)
        load_stageA(x, ew, bm0, t + 2, o, s);
        // Compute on tile t
        compute_tile_ws(As_hi[cur], As_lo[cur], Bs_hi[cur], Bs_lo[cur],
                        lane, wave_m, wave_n, acc);
        wait_asynccnt0();
        __syncthreads();
    }
    // ---- Pipeline tail ----
    {
        const int nxt = (KT - 1) & 1;
        const int cur = (KT - 2) & 1;
        stageA_to_lds(s, o, As_hi[nxt], As_lo[nxt]);
        async_copy_B(Whi + ((size_t)(KT - 1) * NBLK + bn_blk) * 512, bhi_lds[nxt], tid);
        async_copy_B(Wlo + ((size_t)(KT - 1) * NBLK + bn_blk) * 512, blo_lds[nxt], tid);
        compute_tile_ws(As_hi[cur], As_lo[cur], Bs_hi[cur], Bs_lo[cur],
                        lane, wave_m, wave_n, acc);
        wait_asynccnt0();
        __syncthreads();
        compute_tile_ws(As_hi[nxt], As_lo[nxt], Bs_hi[nxt], Bs_lo[nxt],
                        lane, wave_m, wave_n, acc);
    }

    // ---- Output: bias + relu. C/D layout: lane -> N (lane&15), VGPR v -> M ----
    const int hsel = lane >> 4;
    const int coln = lane & 15;
    #pragma unroll
    for (int i = 0; i < 2; ++i) {
        #pragma unroll
        for (int jn = 0; jn < 4; ++jn) {
            const int col = bn0 + wave_n * 64 + jn * 16 + coln;
            const float bv = bias[col];
            #pragma unroll
            for (int v = 0; v < 8; ++v) {
                const int row = bm0 + wave_m * 32 + i * 16 + hsel * 8 + v;
                const float val = acc[i][jn][v] + bv;
                out[row * Ndim + col] = fmaxf(val, 0.0f);
            }
        }
    }
}

// ============================================================================
// Fallback (self-contained, no workspace).
// ============================================================================

struct Stage {
    float4 a[4];
    float  sw[4];
    float2 b0[4];
    float2 b1[4];
};

struct Offsets {
    int aoff[4];
    int eoff[4];
    int boff[4];
    int offA[4];
    int offB[4][2];
};

__device__ __forceinline__ void init_offsets(int tid, Offsets& o) {
    #pragma unroll
    for (int j = 0; j < 4; ++j) {
        const int q  = j * 256 + tid;
        const int r  = q >> 3;
        const int kq = (q & 7) << 2;
        o.aoff[j] = r * Mdim + kq;
        o.eoff[j] = r * Edim;
        const int lnA = (r & 15) | ((kq >> 4) << 4);
        const int hhA = kq & 15;
        o.offA[j] = ((r >> 4) * 512) + lnA * 16 + hhA;
        const int kk = (q >> 6) << 1;
        const int n0 = (q & 63) << 1;
        o.boff[j] = kk * Ndim + n0;
        #pragma unroll
        for (int c = 0; c < 2; ++c) {
            const int n   = n0 + c;
            const int lnB = (n & 15) | ((kk >> 4) << 4);
            o.offB[j][c] = ((n >> 4) * 512) + lnB * 16 + (kk & 15);
        }
    }
}

__device__ __forceinline__ void load_stage(const float* __restrict__ x,
                                           const float* __restrict__ ew,
                                           const float* __restrict__ W,
                                           int bm0, int bn0, int kt,
                                           const Offsets& o, Stage& s)
{
    const int k0 = kt * BK;
    const int e  = k0 >> 10;
    const int m0 = k0 & (Mdim - 1);
    const float* __restrict__ xb  = x  + bm0 * Mdim + m0;
    const float* __restrict__ ewb = ew + bm0 * Edim + e;
    const float* __restrict__ wb  = W  + (long)k0 * Ndim + bn0;
    #pragma unroll
    for (int j = 0; j < 4; ++j) {
        s.a[j]  = *(const float4*)(xb + o.aoff[j]);
        s.sw[j] = ewb[o.eoff[j]];
    }
    #pragma unroll
    for (int j = 0; j < 4; ++j) {
        s.b0[j] = *(const float2*)(wb + o.boff[j]);
        s.b1[j] = *(const float2*)(wb + o.boff[j] + Ndim);
    }
}

__device__ __forceinline__ void stage_to_lds(const Stage& s, const Offsets& o,
                                             unsigned short* __restrict__ As_hi,
                                             unsigned short* __restrict__ As_lo,
                                             unsigned short* __restrict__ Bs_hi,
                                             unsigned short* __restrict__ Bs_lo)
{
    #pragma unroll
    for (int j = 0; j < 4; ++j) {
        const float sw = s.sw[j];
        const float v0 = s.a[j].x * sw, v1 = s.a[j].y * sw;
        const float v2 = s.a[j].z * sw, v3 = s.a[j].w * sw;
        uint2 ph, pl;
        ph.x = pack_hi16(v0, v1);
        ph.y = pack_hi16(v2, v3);
        pl.x = pack_hi16(bf16_rem(v0), bf16_rem(v1));
        pl.y = pack_hi16(bf16_rem(v2), bf16_rem(v3));
        *(uint2*)&As_hi[o.offA[j]] = ph;
        *(uint2*)&As_lo[o.offA[j]] = pl;
    }
    #pragma unroll
    for (int j = 0; j < 4; ++j) {
        const float v0[2] = { s.b0[j].x, s.b0[j].y };
        const float v1[2] = { s.b1[j].x, s.b1[j].y };
        #pragma unroll
        for (int c = 0; c < 2; ++c) {
            *(unsigned int*)&Bs_hi[o.offB[j][c]] = pack_hi16(v0[c], v1[c]);
            *(unsigned int*)&Bs_lo[o.offB[j][c]] = pack_hi16(bf16_rem(v0[c]), bf16_rem(v1[c]));
        }
    }
}

__device__ __forceinline__ void compute_tile(const unsigned short* __restrict__ As_hi,
                                             const unsigned short* __restrict__ As_lo,
                                             const unsigned short* __restrict__ Bs_hi,
                                             const unsigned short* __restrict__ Bs_lo,
                                             int lane, int wave_m, int wave_n,
                                             v8f acc[2][4])
{
    v16bf ah[2], al[2];
    #pragma unroll
    for (int i = 0; i < 2; ++i) {
        const int rb = wave_m * 2 + i;
        ah[i] = *(const v16bf*)&As_hi[rb * 512 + lane * 16];
        al[i] = *(const v16bf*)&As_lo[rb * 512 + lane * 16];
    }
    #pragma unroll
    for (int jn = 0; jn < 4; ++jn) {
        const int cb = wave_n * 4 + jn;
        const v16bf bhi = *(const v16bf*)&Bs_hi[cb * 512 + lane * 16];
        const v16bf blo = *(const v16bf*)&Bs_lo[cb * 512 + lane * 16];
        #pragma unroll
        for (int i = 0; i < 2; ++i) {
            acc[i][jn] = __builtin_amdgcn_wmma_f32_16x16x32_bf16(
                false, ah[i], false, bhi, (short)0, acc[i][jn], false, false);
            acc[i][jn] = __builtin_amdgcn_wmma_f32_16x16x32_bf16(
                false, ah[i], false, blo, (short)0, acc[i][jn], false, false);
            acc[i][jn] = __builtin_amdgcn_wmma_f32_16x16x32_bf16(
                false, al[i], false, bhi, (short)0, acc[i][jn], false, false);
        }
    }
}

__global__ __launch_bounds__(256) void moe_mix_gemm(
    const float* __restrict__ x,
    const float* __restrict__ ew,
    const float* __restrict__ W,
    const float* __restrict__ bias,
    float* __restrict__ out)
{
    __shared__ alignas(32) unsigned short As_hi[2][TILE_A];
    __shared__ alignas(32) unsigned short As_lo[2][TILE_A];
    __shared__ alignas(32) unsigned short Bs_hi[2][TILE_B];
    __shared__ alignas(32) unsigned short Bs_lo[2][TILE_B];

    const int tid    = threadIdx.x;
    const int lane   = tid & 31;
    const int wid    = tid >> 5;
    const int wave_m = wid & 3;
    const int wave_n = wid >> 2;

    const int bm0 = blockIdx.x * BM;
    const int bn0 = blockIdx.y * BN;

    Offsets o;
    init_offsets(tid, o);

    v8f acc[2][4] = {};
    Stage s;

    load_stage(x, ew, W, bm0, bn0, 0, o, s);
    stage_to_lds(s, o, As_hi[0], As_lo[0], Bs_hi[0], Bs_lo[0]);
    load_stage(x, ew, W, bm0, bn0, 1, o, s);
    __syncthreads();

    for (int t = 0; t < KT - 2; ++t) {
        const int nxt = (t + 1) & 1;
        const int cur = t & 1;
        stage_to_lds(s, o, As_hi[nxt], As_lo[nxt], Bs_hi[nxt], Bs_lo[nxt]);
        load_stage(x, ew, W, bm0, bn0, t + 2, o, s);
        compute_tile(As_hi[cur], As_lo[cur], Bs_hi[cur], Bs_lo[cur],
                     lane, wave_m, wave_n, acc);
        __syncthreads();
    }
    {
        const int nxt = (KT - 1) & 1;
        const int cur = (KT - 2) & 1;
        stage_to_lds(s, o, As_hi[nxt], As_lo[nxt], Bs_hi[nxt], Bs_lo[nxt]);
        compute_tile(As_hi[cur], As_lo[cur], Bs_hi[cur], Bs_lo[cur],
                     lane, wave_m, wave_n, acc);
        __syncthreads();
        compute_tile(As_hi[nxt], As_lo[nxt], Bs_hi[nxt], Bs_lo[nxt],
                     lane, wave_m, wave_n, acc);
    }

    const int hsel = lane >> 4;
    const int coln = lane & 15;
    #pragma unroll
    for (int i = 0; i < 2; ++i) {
        #pragma unroll
        for (int jn = 0; jn < 4; ++jn) {
            const int col = bn0 + wave_n * 64 + jn * 16 + coln;
            const float bv = bias[col];
            #pragma unroll
            for (int v = 0; v < 8; ++v) {
                const int row = bm0 + wave_m * 32 + i * 16 + hsel * 8 + v;
                const float val = acc[i][jn][v] + bv;
                out[row * Ndim + col] = fmaxf(val, 0.0f);
            }
        }
    }
}

extern "C" void kernel_launch(void* const* d_in, const int* in_sizes, int n_in,
                              void* d_out, int out_size, void* d_ws, size_t ws_size,
                              hipStream_t stream) {
    (void)in_sizes; (void)n_in; (void)out_size;
    const float* x    = (const float*)d_in[0];
    const float* ew   = (const float*)d_in[1];
    const float* W    = (const float*)d_in[2];
    const float* bias = (const float*)d_in[3];
    float* out        = (float*)d_out;

    const size_t ws_need = (size_t)Kdim * Ndim * 2 * sizeof(unsigned short); // 32 MB
    dim3 grid(Bdim / BM, Ndim / BN);
    if (ws_size >= ws_need) {
        unsigned short* Whi = (unsigned short*)d_ws;
        unsigned short* Wlo = Whi + (size_t)Kdim * Ndim;
        // Pack W into split-bf16 fragment order (runs every call; deterministic).
        pack_w<<<(Kdim / 2) * Ndim / 256, 256, 0, stream>>>(W, Whi, Wlo);
        moe_mix_gemm_ws<<<grid, 256, 0, stream>>>(x, ew, Whi, Wlo, bias, out);
    } else {
        moe_mix_gemm<<<grid, 256, 0, stream>>>(x, ew, W, bias, out);
    }
}
